// AttentionHead_12481174962810
// MI455X (gfx1250) — matrix operand order repaired and checked by reading, not compiled, not run on hardware
//
#include <hip/hip_runtime.h>
#include <hip/hip_bf16.h>

// ---------------- CDNA5 WMMA / TDM types ----------------
typedef __attribute__((ext_vector_type(16))) __bf16 v16bf;
typedef __attribute__((ext_vector_type(8)))  float  v8f;
typedef __attribute__((ext_vector_type(4)))  unsigned int u32x4;
typedef __attribute__((ext_vector_type(8)))  int i32x8;
typedef __attribute__((ext_vector_type(4)))  int i32x4;

union BF16Frag {
    v16bf     v;
    uint4     q[2];
    unsigned  u[8];
};

__device__ __forceinline__ v8f wmma_bf16(v16bf a, v16bf b, v8f c) {
    return __builtin_amdgcn_wmma_f32_16x16x32_bf16(
        false, a, false, b, (short)0, c, false, false);
}

// float -> bf16 round-to-nearest-even
__device__ __forceinline__ unsigned short f2bf(float f) {
    unsigned u = __float_as_uint(f);
    u += 0x7FFFu + ((u >> 16) & 1u);
    return (unsigned short)(u >> 16);
}
__device__ __forceinline__ unsigned pack2(float lo, float hi) {
    return (unsigned)f2bf(lo) | ((unsigned)f2bf(hi) << 16);
}

// 1-D TDM load: n8 contiguous 8-byte units from gsrc -> LDS offset lds_off.
// D# per CDNA5 ISA; 6-arg builtin variant.
__device__ __forceinline__ void tdm_load_1d(unsigned lds_off, const void* gsrc,
                                            unsigned n8) {
    unsigned long long ga = (unsigned long long)(size_t)gsrc;
    u32x4 g0;
    g0[0] = 1u;                                   // count=1 (valid, user mode)
    g0[1] = lds_off;                              // lds_addr (bytes)
    g0[2] = (unsigned)ga;                         // global_addr[31:0]
    g0[3] = (unsigned)((ga >> 32) & 0x01FFFFFFu)  // global_addr[56:32]
            | 0x80000000u;                        // type=2 ("image")
    i32x8 g1;
    g1[0] = (int)(3u << 16);                      // data_size = 8 bytes
    g1[1] = (int)((n8 & 0xFFFFu) << 16);          // tensor_dim0[15:0]
    g1[2] = (int)((n8 >> 16) | (1u << 16));       // tensor_dim0 hi | tensor_dim1=1
    g1[3] = (int)(n8 << 16);                      // tensor_dim1 hi=0 | tile_dim0=n8
    g1[4] = 1;                                    // tile_dim1=1, tile_dim2=0
    g1[5] = (int)n8;                              // tensor_dim0_stride lo
    g1[6] = 0;
    g1[7] = 0;
    i32x4 z4 = (i32x4){0, 0, 0, 0};
    i32x8 z8 = (i32x8){0, 0, 0, 0, 0, 0, 0, 0};
    __builtin_amdgcn_tensor_load_to_lds(g0, g1, z4, z4, z8, 0);
}

// ---------------- problem constants ----------------
#define BATCH 8
#define TSEQ  4096
#define EDIM  1024
#define DDIM  64
#define CKS   4                      // k-steps per TDM chunk
#define KSTEP_ELEMS 6144             // bf16 elems per k-step of packed W (3*4*512)
#define CHUNK_ELEMS (CKS * KSTEP_ELEMS)

// workspace layout (bytes)
#define WP_OFF  0u                       // 3*E*D bf16 = 393216
#define QT_OFF  (WP_OFF + 393216u)       // B*D*T bf16 (Q transposed (D,T))
#define KT_OFF  (QT_OFF + 4194304u)      // B*D*T bf16 (K transposed (D,T))
#define VB_OFF  (KT_OFF + 4194304u)      // B*T*D bf16 (V row-major)
#define RB_OFF  (VB_OFF + 4194304u)      // B*64*64 bf16 (softmax(R)^T)
#define RP_OFF  (RB_OFF + 65536u)        // 8*B*64*64 f32 score partials

// =====================================================================
// Kernel 1: repack Wq/Wk/Wv into WMMA B-fragment layout, bf16.
// Layout: [ks][m][nt][lane][j];  n = nt*16+(lane&15); k = ks*32+(lane>>4)*8+(j&7)+(j>>3)*16
// =====================================================================
__global__ void pack_w_kernel(const float* __restrict__ Wq,
                              const float* __restrict__ Wk,
                              const float* __restrict__ Wv,
                              unsigned short* __restrict__ wp) {
    int idx = blockIdx.x * 256 + threadIdx.x;
    if (idx >= 3 * EDIM * DDIM) return;
    int j  = idx & 15;
    int l  = (idx >> 4) & 31;
    int t  = idx >> 9;
    int nt = t & 3;
    int u  = t >> 2;
    int m  = u % 3;
    int ks = u / 3;
    int n = nt * 16 + (l & 15);
    int k = ks * 32 + (l >> 4) * 8 + (j & 7) + (j >> 3) * 16;
    const float* W = (m == 0) ? Wq : ((m == 1) ? Wk : Wv);
    wp[idx] = f2bf(W[k * DDIM + n]);
}

__device__ __forceinline__ void ldB(BF16Frag& B, const unsigned short* base,
                                    int tile, int lane) {
    const uint4* bp = (const uint4*)(base + tile * 512 + lane * 16);
    B.q[0] = bp[0];
    B.q[1] = bp[1];
}

// =====================================================================
// Kernel 2: fused QKV projection, TDM-staged weights.
// Per chunk (4 k-steps): wave0 TDM-prefetches next 48KB W chunk while all
// waves run 48 WMMAs off LDS with a depth-2 rotating B-fragment pipeline;
// x double-buffered in registers.
// =====================================================================
__global__ __launch_bounds__(256) void qkv_kernel(
        const float* __restrict__ x, const unsigned short* __restrict__ wp,
        unsigned short* __restrict__ QT, unsigned short* __restrict__ KT,
        unsigned short* __restrict__ Vb) {
    __shared__ unsigned short Wlds[2][CHUNK_ELEMS];   // 2 x 48KB

    const int wave = threadIdx.x >> 5;
    const int lane = threadIdx.x & 31;
    const int b  = blockIdx.x >> 5;
    const int tb = (blockIdx.x & 31) * 128 + wave * 16;

    const int arow  = tb + (lane & 15);
    const int khalf = (lane >> 4) ? 8 : 0;
    const float* xrow = x + ((long)b * TSEQ + arow) * EDIM;

    v8f acc[12];
#pragma unroll
    for (int i = 0; i < 12; ++i) acc[i] = (v8f){0,0,0,0,0,0,0,0};

    // prologue: TDM chunk 0, and raw x for k-step 0
    if (wave == 0)
        tdm_load_1d((unsigned)(size_t)&Wlds[0][0], wp, CHUNK_ELEMS * 2 / 8);
    float4 ra0 = *(const float4*)(xrow + khalf);
    float4 ra1 = *(const float4*)(xrow + khalf + 4);
    float4 ra2 = *(const float4*)(xrow + khalf + 16);
    float4 ra3 = *(const float4*)(xrow + khalf + 20);

    const int NCH = (EDIM / 32) / CKS;   // 8 chunks
    for (int c = 0; c < NCH; ++c) {
        if (wave == 0) __builtin_amdgcn_s_wait_tensorcnt(0);
        __syncthreads();   // chunk c visible; everyone done reading chunk c-1
        if (wave == 0 && c + 1 < NCH)
            tdm_load_1d((unsigned)(size_t)&Wlds[(c + 1) & 1][0],
                        wp + (long)(c + 1) * CHUNK_ELEMS, CHUNK_ELEMS * 2 / 8);

        const unsigned short* wb = &Wlds[c & 1][0];
#pragma unroll
        for (int ki = 0; ki < CKS; ++ki) {
            const int s = c * CKS + ki;
            // prefetch next k-step of x while this step computes
            float4 rn0, rn1, rn2, rn3;
            if (s + 1 < EDIM / 32) {
                const float* xn = xrow + (s + 1) * 32 + khalf;
                rn0 = *(const float4*)(xn);
                rn1 = *(const float4*)(xn + 4);
                rn2 = *(const float4*)(xn + 16);
                rn3 = *(const float4*)(xn + 20);
            }
            BF16Frag A;
            A.u[0] = pack2(ra0.x, ra0.y); A.u[1] = pack2(ra0.z, ra0.w);
            A.u[2] = pack2(ra1.x, ra1.y); A.u[3] = pack2(ra1.z, ra1.w);
            A.u[4] = pack2(ra2.x, ra2.y); A.u[5] = pack2(ra2.z, ra2.w);
            A.u[6] = pack2(ra3.x, ra3.y); A.u[7] = pack2(ra3.z, ra3.w);

            const unsigned short* wk = wb + ki * KSTEP_ELEMS;
            // depth-2 rotating B-fragment pipeline over the 12 tiles
            BF16Frag Bf[3];
            ldB(Bf[0], wk, 0, lane);
            ldB(Bf[1], wk, 1, lane);
#pragma unroll
            for (int i = 0; i < 12; ++i) {
                if (i + 2 < 12) ldB(Bf[(i + 2) % 3], wk, i + 2, lane);
                acc[i] = wmma_bf16(A.v, Bf[i % 3].v, acc[i]);
            }
            ra0 = rn0; ra1 = rn1; ra2 = rn2; ra3 = rn3;
        }
    }

    // C layout: row m = v + 8*(lane>=16), col n = nt*16 + (lane&15)
    const int crow = tb + ((lane >> 4) ? 8 : 0);
#pragma unroll
    for (int nt = 0; nt < 4; ++nt) {
        const int ncol = nt * 16 + (lane & 15);
#pragma unroll
        for (int v = 0; v < 8; ++v) {
            const int t = crow + v;
            QT[((long)b * DDIM + ncol) * TSEQ + t] = f2bf(acc[0 * 4 + nt][v]);
            KT[((long)b * DDIM + ncol) * TSEQ + t] = f2bf(acc[1 * 4 + nt][v]);
            Vb[((long)b * TSEQ + t) * DDIM + ncol] = f2bf(acc[2 * 4 + nt][v]);
        }
    }
}

// =====================================================================
// Kernel 3a: partial scores.  Grid: B*8 blocks; block (b,slice) reduces a
// T-slice of 512 (16 k-steps), register double-buffered, writes 64x64 f32.
// =====================================================================
__global__ __launch_bounds__(256) void scores_partial_kernel(
        const unsigned short* __restrict__ QT,
        const unsigned short* __restrict__ KT,
        float* __restrict__ Rp) {
    const int b    = blockIdx.x >> 3;
    const int sl   = blockIdx.x & 7;
    const int wave = threadIdx.x >> 5;
    const int lane = threadIdx.x & 31;
    const int dt   = wave & 3;
    const int et0  = (wave >> 2) * 2;

    const unsigned short* Kt = KT + (long)b * DDIM * TSEQ;
    const unsigned short* Qt = QT + (long)b * DDIM * TSEQ;

    const int arow  = dt * 16 + (lane & 15);
    const int akoff = (lane >> 4) * 16;
    const int bkoff = (lane >> 4) * 16;
    const int brow0 = et0 * 16 + (lane & 15);

    const unsigned short* ka = Kt + (long)arow * TSEQ + sl * 512 + akoff;
    const unsigned short* q0 = Qt + (long)brow0 * TSEQ + sl * 512 + bkoff;
    const unsigned short* q1 = q0 + 16l * TSEQ;

    v8f acc0 = (v8f){0,0,0,0,0,0,0,0};
    v8f acc1 = (v8f){0,0,0,0,0,0,0,0};

    // current raw data (k-step 0)
    uint4 ca0 = ((const uint4*)ka)[0];
    uint4 ca1 = ((const uint4*)(ka + 8))[0];
    uint4 cb00 = ((const uint4*)q0)[0], cb01 = ((const uint4*)q0)[1];
    uint4 cb10 = ((const uint4*)q1)[0], cb11 = ((const uint4*)q1)[1];

#pragma unroll
    for (int ks = 0; ks < 16; ++ks) {
        uint4 na0, na1, nb00, nb01, nb10, nb11;
        if (ks + 1 < 16) {
            const int kb = (ks + 1) * 32;
            na0  = ((const uint4*)(ka + kb))[0];
            na1  = ((const uint4*)(ka + kb + 8))[0];
            nb00 = ((const uint4*)(q0 + kb))[0];
            nb01 = ((const uint4*)(q0 + kb))[1];
            nb10 = ((const uint4*)(q1 + kb))[0];
            nb11 = ((const uint4*)(q1 + kb))[1];
        }
        BF16Frag A;  A.q[0]  = ca0;  A.q[1]  = ca1;
        BF16Frag B0; B0.q[0] = cb00; B0.q[1] = cb01;
        BF16Frag B1; B1.q[0] = cb10; B1.q[1] = cb11;
        acc0 = wmma_bf16(A.v, B0.v, acc0);
        acc1 = wmma_bf16(A.v, B1.v, acc1);
        ca0 = na0; ca1 = na1;
        cb00 = nb00; cb01 = nb01; cb10 = nb10; cb11 = nb11;
    }

    float* out = Rp + (long)blockIdx.x * 4096;
    const int crow = dt * 16 + ((lane >> 4) ? 8 : 0);
    const int ccol = et0 * 16 + (lane & 15);
#pragma unroll
    for (int v = 0; v < 8; ++v) {
        out[(crow + v) * DDIM + ccol]      = acc0[v];
        out[(crow + v) * DDIM + ccol + 16] = acc1[v];
    }
}

// =====================================================================
// Kernel 3b: finalize scores. Sum 8 partials (fixed order -> deterministic),
// scale, exact reference mask (j<i || R[j][i]==0), row softmax, store
// transposed bf16: Rb[b][e*64+dd] = softmax(R)[dd][e].
// =====================================================================
__global__ __launch_bounds__(256) void softmax_kernel(
        const float* __restrict__ Rp, unsigned short* __restrict__ Rb) {
    __shared__ float R [DDIM * DDIM];
    __shared__ float Rm[DDIM * DDIM];
    const int b = blockIdx.x;
    const float scale = 0.03125f;  // 1/sqrt(1024)

#pragma unroll
    for (int it = 0; it < 16; ++it) {
        const int idx = it * 256 + threadIdx.x;
        float s = 0.0f;
#pragma unroll
        for (int sl = 0; sl < 8; ++sl)
            s += Rp[((long)b * 8 + sl) * 4096 + idx];
        R[idx] = s * scale;
    }
    __syncthreads();

    const int tx = threadIdx.x & 63;
    const int ty = threadIdx.x >> 6;
#pragma unroll
    for (int rr = 0; rr < 16; ++rr) {
        const int i = rr * 4 + ty, j = tx;
        const float val = R[i * DDIM + j];
        const bool  msk = (j < i) || (R[j * DDIM + i] == 0.0f);
        Rm[i * DDIM + j] = msk ? -__builtin_inff() : val;
    }
    __syncthreads();

    if (threadIdx.x < DDIM) {
        const int i = threadIdx.x;
        float mx = -__builtin_inff();
        for (int j = 0; j < DDIM; ++j) mx = fmaxf(mx, Rm[i * DDIM + j]);
        float s = 0.0f;
        for (int j = 0; j < DDIM; ++j) {
            float e = __expf(Rm[i * DDIM + j] - mx);
            Rm[i * DDIM + j] = e;
            s += e;
        }
        const float inv = 1.0f / s;
        for (int j = 0; j < DDIM; ++j) Rm[i * DDIM + j] *= inv;
    }
    __syncthreads();

#pragma unroll
    for (int it = 0; it < 16; ++it) {
        const int idx = it * 256 + threadIdx.x;
        const int e = idx >> 6, dd = idx & 63;
        Rb[(long)b * 4096 + idx] = f2bf(Rm[dd * DDIM + e]);
    }
}

// =====================================================================
// Kernel 4: out = V @ R.  Grid: B*T/128 blocks x 8 waves; wave: 16 rows x
// 64 cols (4 tiles), K=64 -> 2 k-steps -> 8 WMMAs.
// =====================================================================
__global__ __launch_bounds__(256) void out_kernel(
        const unsigned short* __restrict__ Vb,
        const unsigned short* __restrict__ Rb,
        float* __restrict__ out) {
    const int wave = threadIdx.x >> 5;
    const int lane = threadIdx.x & 31;
    const int b  = blockIdx.x >> 5;
    const int tb = (blockIdx.x & 31) * 128 + wave * 16;

    const unsigned short* Vp = Vb + (long)b * TSEQ * DDIM;
    const unsigned short* Rp = Rb + (long)b * 4096;

    const int arow  = tb + (lane & 15);
    const int akoff = (lane >> 4) * 16;
    const int bkoff = (lane >> 4) * 16;

    v8f acc[4];
#pragma unroll
    for (int i = 0; i < 4; ++i) acc[i] = (v8f){0,0,0,0,0,0,0,0};

#pragma unroll
    for (int ks = 0; ks < 2; ++ks) {
        const int kb = ks * 32;
        BF16Frag A;
        A.q[0] = *(const uint4*)(Vp + (long)arow * DDIM + kb + akoff);
        A.q[1] = *(const uint4*)(Vp + (long)arow * DDIM + kb + akoff + 8);
#pragma unroll
        for (int nt = 0; nt < 4; ++nt) {
            const int brow = nt * 16 + (lane & 15);
            const uint4* bp = (const uint4*)(Rp + (long)brow * DDIM + kb + bkoff);
            BF16Frag B; B.q[0] = bp[0]; B.q[1] = bp[1];
            acc[nt] = wmma_bf16(A.v, B.v, acc[nt]);
        }
    }

    const int crow = tb + ((lane >> 4) ? 8 : 0);
#pragma unroll
    for (int nt = 0; nt < 4; ++nt) {
        const int ncol = nt * 16 + (lane & 15);
#pragma unroll
        for (int v = 0; v < 8; ++v)
            out[((long)b * TSEQ + crow + v) * DDIM + ncol] = acc[nt][v];
    }
}

// =====================================================================
extern "C" void kernel_launch(void* const* d_in, const int* in_sizes, int n_in,
                              void* d_out, int out_size, void* d_ws, size_t ws_size,
                              hipStream_t stream) {
    const float* x  = (const float*)d_in[0];
    const float* Wq = (const float*)d_in[1];
    const float* Wk = (const float*)d_in[2];
    const float* Wv = (const float*)d_in[3];
    float* out = (float*)d_out;

    char* ws = (char*)d_ws;
    unsigned short* wp = (unsigned short*)(ws + WP_OFF);
    unsigned short* QT = (unsigned short*)(ws + QT_OFF);
    unsigned short* KT = (unsigned short*)(ws + KT_OFF);
    unsigned short* Vb = (unsigned short*)(ws + VB_OFF);
    unsigned short* Rb = (unsigned short*)(ws + RB_OFF);
    float*          Rp = (float*)(ws + RP_OFF);

    pack_w_kernel<<<(3 * EDIM * DDIM + 255) / 256, 256, 0, stream>>>(Wq, Wk, Wv, wp);
    qkv_kernel<<<BATCH * TSEQ / 128, 256, 0, stream>>>(x, wp, QT, KT, Vb);
    scores_partial_kernel<<<BATCH * 8, 256, 0, stream>>>(QT, KT, Rp);
    softmax_kernel<<<BATCH, 256, 0, stream>>>(Rp, Rb);
    out_kernel<<<BATCH * TSEQ / 128, 256, 0, stream>>>(Vb, Rb, out);
}
